// SelfAttention_82626580840878
// MI455X (gfx1250) — matrix operand order repaired and checked by reading, not compiled, not run on hardware
//
#include <hip/hip_runtime.h>
#include <hip/hip_bf16.h>

typedef __attribute__((ext_vector_type(16))) _Float16 v16h;
typedef __attribute__((ext_vector_type(8)))  _Float16 v8h;
typedef __attribute__((ext_vector_type(4)))  _Float16 v4h;
typedef __attribute__((ext_vector_type(8)))  float    v8f;

#define NB 8
#define NS 2048
#define ND 512

static __device__ __forceinline__ v8f wmma_f16(v16h a, v16h b, v8f c) {
    return __builtin_amdgcn_wmma_f32_16x16x32_f16(
        /*neg_a=*/false, a, /*neg_b=*/false, b,
        /*c_mod=*/(short)0, c, /*reuse_a=*/false, /*reuse_b=*/false);
}

// CDNA5 async global->LDS copy, 16 bytes, tracked by ASYNCcnt.
// generic->LDS address is the low 32 bits (aperture truncation).
static __device__ __forceinline__ void async_b128(void* lds, const void* g) {
    asm volatile("global_load_async_to_lds_b128 %0, %1, off"
                 :: "v"((unsigned int)(unsigned long long)lds), "v"(g)
                 : "memory");
}
static __device__ __forceinline__ void wait_async0() {
    asm volatile("s_wait_asynccnt 0x0" ::: "memory");
}

// A-fragment (16x32 f16, M x K): lanes 0-15 row M=lane, K in {0..7,16..23};
// lanes 16-31 row M=lane-16, K in {8..15,24..31}. Two contiguous 16B loads.
static __device__ __forceinline__ v16h ld_frag_a(const _Float16* p, int lda, int lane) {
    const _Float16* q = p + (size_t)(lane & 15) * lda + ((lane >> 4) << 3);
    v8h lo = *(const v8h*)q;
    v8h hi = *(const v8h*)(q + 16);
    return __builtin_shufflevector(lo, hi, 0,1,2,3,4,5,6,7,8,9,10,11,12,13,14,15);
}

// B-fragment (32x16 f16, K x N) loaded from B^T rows ([N][K] memory):
// lane = N column; lanes 0-15 hold K {0..7,16..23}, lanes 16-31 hold K {8..15,24..31}.
static __device__ __forceinline__ v16h ld_frag_b(const _Float16* p, int ldb, int lane) {
    const _Float16* q = p + (size_t)(lane & 15) * ldb + ((lane >> 4) << 3);
    v8h lo = *(const v8h*)q;
    v8h hi = *(const v8h*)(q + 16);
    return __builtin_shufflevector(lo, hi, 0,1,2,3,4,5,6,7,8,9,10,11,12,13,14,15);
}

// ---------------------------------------------------------------- conversions
__global__ void cvt_f32_to_f16(const float* __restrict__ in,
                               _Float16* __restrict__ out, size_t n4) {
    size_t i = (size_t)blockIdx.x * blockDim.x + threadIdx.x;
    size_t stride = (size_t)gridDim.x * blockDim.x;
    const float4* in4 = (const float4*)in;
    v4h* out4 = (v4h*)out;
    for (; i < n4; i += stride) {
        float4 v = in4[i];
        v4h o;
        o.x = (_Float16)v.x; o.y = (_Float16)v.y;
        o.z = (_Float16)v.z; o.w = (_Float16)v.w;
        out4[i] = o;
    }
}

// W [K=512][N=512] f32 -> Wt [N][K] f16
__global__ void transpose_cvt(const float* __restrict__ W, _Float16* __restrict__ Wt) {
    __shared__ float t[16][17];
    int k = blockIdx.y * 16 + threadIdx.y;
    int n = blockIdx.x * 16 + threadIdx.x;
    t[threadIdx.y][threadIdx.x] = W[(size_t)k * ND + n];
    __syncthreads();
    int nn = blockIdx.x * 16 + threadIdx.y;
    int kk = blockIdx.y * 16 + threadIdx.x;
    Wt[(size_t)nn * ND + kk] = (_Float16)t[threadIdx.x][threadIdx.y];
}

// ---------------------------------------------------------------- WMMA GEMM core
// acc[2][4] = A-tile(128xK) @ BT-tile(128xK)^T slice for this wave.
// Double-buffered LDS, async global->LDS staging, one barrier per K-step.
static __device__ __forceinline__ void gemm_core(
    const _Float16* __restrict__ A,   // [M][512] tile origin row m0
    const _Float16* __restrict__ BT,  // [512][512] tile origin row n0
    int m0, int n0, int tid, int lane, int wr, int wc, v8f acc[2][4]) {
    constexpr int K = ND;
    __shared__ __align__(16) _Float16 As[2][128][40];
    __shared__ __align__(16) _Float16 Bs[2][128][40];

    const int r  = tid >> 2;            // 0..63
    const int cc = (tid & 3) * 8;       // 0,8,16,24 halves
    const _Float16* ga0 = A  + (size_t)m0 * K;
    const _Float16* gb0 = BT + (size_t)n0 * K;

    // prefetch tile 0
    {
        const _Float16* ga = ga0;
        const _Float16* gb = gb0;
        async_b128(&As[0][r][cc],      ga + (size_t)r        * K + cc);
        async_b128(&As[0][r + 64][cc], ga + (size_t)(r + 64) * K + cc);
        async_b128(&Bs[0][r][cc],      gb + (size_t)r        * K + cc);
        async_b128(&Bs[0][r + 64][cc], gb + (size_t)(r + 64) * K + cc);
    }

    for (int k0 = 0; k0 < K; k0 += 32) {
        const int cur = (k0 >> 5) & 1;
        wait_async0();                  // this wave's copies (into cur) landed
        __syncthreads();                // => everyone's copies landed, prev reads done
        if (k0 + 32 < K) {              // stage next tile into the other buffer
            const _Float16* ga = ga0 + k0 + 32;
            const _Float16* gb = gb0 + k0 + 32;
            async_b128(&As[cur ^ 1][r][cc],      ga + (size_t)r        * K + cc);
            async_b128(&As[cur ^ 1][r + 64][cc], ga + (size_t)(r + 64) * K + cc);
            async_b128(&Bs[cur ^ 1][r][cc],      gb + (size_t)r        * K + cc);
            async_b128(&Bs[cur ^ 1][r + 64][cc], gb + (size_t)(r + 64) * K + cc);
        }
        v16h af[2], bf[4];
        af[0] = ld_frag_a(&As[cur][wr * 32][0], 40, lane);
        af[1] = ld_frag_a(&As[cur][wr * 32 + 16][0], 40, lane);
#pragma unroll
        for (int nt = 0; nt < 4; ++nt)
            bf[nt] = ld_frag_b(&Bs[cur][wc * 64 + nt * 16][0], 40, lane);
#pragma unroll
        for (int mi = 0; mi < 2; ++mi)
#pragma unroll
            for (int nt = 0; nt < 4; ++nt)
                acc[mi][nt] = wmma_f16(af[mi], bf[nt], acc[mi][nt]);
    }
}

// C[M=16384][N=512] = A[M][K=512] @ Wt^T, f16 out.
// MODE 0: plain. MODE 1: scaled (q). MODE 2: transposed per-batch [B][N][S] (v).
template <int MODE>
__global__ __launch_bounds__(256) void gemm_rcr_f16(
    const _Float16* __restrict__ A, const _Float16* __restrict__ BT,
    _Float16* __restrict__ C, float scale) {
    constexpr int N = ND;
    const int tid = threadIdx.x, lane = tid & 31, w = tid >> 5;
    const int wr = w >> 1, wc = w & 1;
    const int m0 = blockIdx.y * 128, n0 = blockIdx.x * 128;
    const int hi16 = (lane >> 4) << 3;

    v8f acc[2][4];
    for (int i = 0; i < 2; ++i)
        for (int j = 0; j < 4; ++j) acc[i][j] = (v8f)(0.0f);

    gemm_core(A, BT, m0, n0, tid, lane, wr, wc, acc);

    for (int mi = 0; mi < 2; ++mi) {
        const int bm = m0 + wr * 32 + mi * 16 + hi16;   // base row for i=0
        for (int nt = 0; nt < 4; ++nt) {
            const int n = n0 + wc * 64 + nt * 16 + (lane & 15);
            if (MODE == 2) {
                const int bb = bm >> 11;                // / 2048
                const int s  = bm & (NS - 1);
                v8h pk;
                for (int i = 0; i < 8; ++i) pk[i] = (_Float16)acc[mi][nt][i];
                *(v8h*)&C[((size_t)bb * ND + n) * NS + s] = pk;
            } else {
                const float sc = (MODE == 1) ? scale : 1.0f;
                for (int i = 0; i < 8; ++i)
                    C[(size_t)(bm + i) * N + n] = (_Float16)(acc[mi][nt][i] * sc);
            }
        }
    }
}

// out[M][512] f32 = A[M][512] @ Wot^T + bo
__global__ __launch_bounds__(256) void gemm_out_f32(
    const _Float16* __restrict__ A, const _Float16* __restrict__ BT,
    const float* __restrict__ bo, float* __restrict__ C) {
    constexpr int N = ND;
    const int tid = threadIdx.x, lane = tid & 31, w = tid >> 5;
    const int wr = w >> 1, wc = w & 1;
    const int m0 = blockIdx.y * 128, n0 = blockIdx.x * 128;
    const int hi16 = (lane >> 4) << 3;

    v8f acc[2][4];
    for (int i = 0; i < 2; ++i)
        for (int j = 0; j < 4; ++j) acc[i][j] = (v8f)(0.0f);

    gemm_core(A, BT, m0, n0, tid, lane, wr, wc, acc);

    for (int mi = 0; mi < 2; ++mi) {
        const int bm = m0 + wr * 32 + mi * 16 + hi16;
        for (int nt = 0; nt < 4; ++nt) {
            const int n = n0 + wc * 64 + nt * 16 + (lane & 15);
            const float b = bo[n];
            for (int i = 0; i < 8; ++i)
                C[(size_t)(bm + i) * N + n] = acc[mi][nt][i] + b;
        }
    }
}

// ---------------------------------------------------------------- flash attention
// One block = (batch bb, 16 query rows). 8 waves: each owns 16 keys of a
// 128-key chunk for scores, and a 64-wide D-slice for the PV accumulation.
__global__ __launch_bounds__(256) void flash_attn(
    const _Float16* __restrict__ qh,   // [B*S][512], pre-scaled by 1/sqrt(D)
    const _Float16* __restrict__ kh,   // [B*S][512]
    const _Float16* __restrict__ vT,   // [B][512][2048]
    _Float16* __restrict__ oh) {       // [B*S][512]
    const int q0 = blockIdx.x * 16;
    const int bb = blockIdx.y;
    const int tid = threadIdx.x, lane = tid & 31, w = tid >> 5;
    const int hi16 = (lane >> 4) << 3;

    __shared__ __align__(16) _Float16 qs[16][512];
    __shared__ __align__(16) float    ss[16][136];
    __shared__ __align__(16) _Float16 ps[16][136];
    __shared__ float mrow[16], lrow[16], arow[16];

    {   // stage the 16x512 query tile via async copies
        int row = tid >> 4;
        int col = (tid & 15) * 32;
        const _Float16* src = qh + ((size_t)(bb * NS + q0 + row)) * ND + col;
        async_b128(&qs[row][col],      src);
        async_b128(&qs[row][col + 8],  src + 8);
        async_b128(&qs[row][col + 16], src + 16);
        async_b128(&qs[row][col + 24], src + 24);
    }
    if (tid < 16) { mrow[tid] = -1e30f; lrow[tid] = 0.0f; }
    wait_async0();

    v8f oacc[4];
    for (int i = 0; i < 4; ++i) oacc[i] = (v8f)(0.0f);
    const int d0 = w * 64;

    for (int kc = 0; kc < NS; kc += 128) {
        __syncthreads();
        // --- scores for this wave's 16 keys: S = q @ k^T (K dim = D) ---
        v8f sacc = (v8f)(0.0f);
        const _Float16* kbase = kh + ((size_t)(bb * NS + kc + w * 16)) * ND;
#pragma unroll 4
        for (int kk = 0; kk < ND; kk += 32) {
            v16h a  = ld_frag_a(&qs[0][kk], 512, lane);
            v16h bf = ld_frag_b(kbase + kk, ND, lane);
            sacc = wmma_f16(a, bf, sacc);
        }
        {
            int col = w * 16 + (lane & 15);
            for (int i = 0; i < 8; ++i) ss[i + hi16][col] = sacc[i];
        }
        __syncthreads();
        // --- online softmax over the 128-key chunk (wave 0) ---
        if (w == 0 && lane < 16) {
            const int r = lane;
            float mprev = mrow[r];
            float cmax = -1e30f;
            for (int j = 0; j < 128; ++j) cmax = fmaxf(cmax, ss[r][j]);
            float mnew = fmaxf(mprev, cmax);
            float al = __expf(mprev - mnew);
            float sum = 0.0f;
            for (int j = 0; j < 128; ++j) {
                float p = __expf(ss[r][j] - mnew);
                ps[r][j] = (_Float16)p;
                sum += p;
            }
            mrow[r] = mnew;
            lrow[r] = lrow[r] * al + sum;
            arow[r] = al;
        }
        __syncthreads();
        // --- rescale O, then O += P @ V-slice ---
        for (int i = 0; i < 8; ++i) {
            float al = arow[i + hi16];
            for (int nt = 0; nt < 4; ++nt) oacc[nt][i] *= al;
        }
#pragma unroll
        for (int kk = 0; kk < 128; kk += 32) {
            v16h a = ld_frag_a(&ps[0][kk], 136, lane);
#pragma unroll
            for (int nt = 0; nt < 4; ++nt) {
                const _Float16* vbase =
                    vT + ((size_t)bb * ND + d0 + nt * 16) * (size_t)NS + kc + kk;
                v16h bf = ld_frag_b(vbase, NS, lane);
                oacc[nt] = wmma_f16(a, bf, oacc[nt]);
            }
        }
    }
    __syncthreads();
    for (int i = 0; i < 8; ++i) {
        float inv = 1.0f / lrow[i + hi16];
        int m = q0 + i + hi16;
        for (int nt = 0; nt < 4; ++nt) {
            int n = d0 + nt * 16 + (lane & 15);
            oh[((size_t)(bb * NS + m)) * ND + n] = (_Float16)(oacc[nt][i] * inv);
        }
    }
}

// ---------------------------------------------------------------- launch
extern "C" void kernel_launch(void* const* d_in, const int* in_sizes, int n_in,
                              void* d_out, int out_size, void* d_ws, size_t ws_size,
                              hipStream_t stream) {
    const float* x  = (const float*)d_in[0];
    const float* Wv = (const float*)d_in[1];
    const float* Wk = (const float*)d_in[2];
    const float* Wq = (const float*)d_in[3];
    const float* Wo = (const float*)d_in[4];
    const float* bo = (const float*)d_in[5];
    float* out = (float*)d_out;

    char* ws = (char*)d_ws;
    const size_t MB = 1ull << 20;
    _Float16* xh  = (_Float16*)(ws);             // 16 MB
    _Float16* qh  = (_Float16*)(ws + 16 * MB);   // 16 MB
    _Float16* kh  = (_Float16*)(ws + 32 * MB);   // 16 MB
    _Float16* vT  = (_Float16*)(ws + 48 * MB);   // 16 MB, [B][D][S]
    _Float16* oh  = (_Float16*)(ws + 64 * MB);   // 16 MB
    _Float16* wqt = (_Float16*)(ws + 80 * MB);   // 512 KB each
    _Float16* wkt = (_Float16*)(ws + 80 * MB + 512 * 1024);
    _Float16* wvt = (_Float16*)(ws + 81 * MB);
    _Float16* wot = (_Float16*)(ws + 81 * MB + 512 * 1024);

    const size_t nx = (size_t)NB * NS * ND;      // 8,388,608
    cvt_f32_to_f16<<<4096, 256, 0, stream>>>(x, xh, nx / 4);

    dim3 tgrid(32, 32), tblk(16, 16);
    transpose_cvt<<<tgrid, tblk, 0, stream>>>(Wq, wqt);
    transpose_cvt<<<tgrid, tblk, 0, stream>>>(Wk, wkt);
    transpose_cvt<<<tgrid, tblk, 0, stream>>>(Wv, wvt);
    transpose_cvt<<<tgrid, tblk, 0, stream>>>(Wo, wot);

    const float scale = 0.044194173824159216f;   // 1/sqrt(512)
    dim3 ggrid(4, 128);                          // N/128 x M/128, M = 16384
    gemm_rcr_f16<1><<<ggrid, 256, 0, stream>>>(xh, wqt, qh, scale);
    gemm_rcr_f16<0><<<ggrid, 256, 0, stream>>>(xh, wkt, kh, 1.0f);
    gemm_rcr_f16<2><<<ggrid, 256, 0, stream>>>(xh, wvt, vT, 1.0f);

    dim3 fgrid(NS / 16, NB);                     // 128 x 8
    flash_attn<<<fgrid, 256, 0, stream>>>(qh, kh, vT, oh);

    gemm_out_f32<<<ggrid, 256, 0, stream>>>(oh, wot, bo, out);
}
